// BitcoinPredictor_24644522344835
// MI455X (gfx1250) — matrix operand order repaired and checked by reading, not compiled, not run on hardware
//
#include <hip/hip_runtime.h>

typedef __attribute__((ext_vector_type(16))) __bf16 v16bf;
typedef __attribute__((ext_vector_type(8)))  float  v8f;
typedef __attribute__((ext_vector_type(4)))  float  v4f;
typedef __attribute__((ext_vector_type(4)))  unsigned int v4u;

#define T_   2048
#define IN_  16
#define H_   128
#define NG_  512   // 4*H

// gfx1250 V_TANH_F32 (single TRANS op); fallback for safety.
#if __has_builtin(__builtin_amdgcn_tanhf)
__device__ __forceinline__ float fast_tanh(float x) { return __builtin_amdgcn_tanhf(x); }
#else
__device__ __forceinline__ float fast_tanh(float x) {
    float e = __expf(2.0f * x);
    return 1.0f - 2.0f / (e + 1.0f);
}
#endif
__device__ __forceinline__ float fast_sigm(float x) {
    return fmaf(0.5f, fast_tanh(0.5f * x), 0.5f);   // sigmoid(x) = 0.5*tanh(x/2)+0.5
}

// One workgroup = 4 waves = 1 per SIMD of a WGP; covers a 16-row batch tile.
// Wave q computes gates (i,f,g,o) for hidden slices {2q, 2q+1}; all weight
// B-fragments are preloaded into registers (gfx1250 MSB VGPR space), so the
// steady-state loop touches LDS only for the h ping-pong staging exchange.
__global__ __launch_bounds__(128, 1)
void slstm_wmma_kernel(const float* __restrict__ x,
                       const float* __restrict__ W_ih,
                       const float* __restrict__ W_hh,
                       const float* __restrict__ b_ih,
                       const float* __restrict__ b_hh,
                       const float* __restrict__ fc_w,
                       const float* __restrict__ fc_b,
                       float* __restrict__ out)
{
    __shared__ __bf16 Hst[2][16][H_];   // ping-pong h staging, 8 KB

    const int tid  = threadIdx.x;
    const int wq   = tid >> 5;        // wave 0..3 -> hidden slices 2*wq, 2*wq+1
    const int lane = tid & 31;
    const int ln16 = lane & 15;
    const int half = lane >> 4;

    {   // zero both h staging buffers (h0 = 0)
        unsigned int* p = (unsigned int*)Hst;
        for (int i = tid; i < 2 * 16 * H_ / 2; i += 128) p[i] = 0u;
    }
    __syncthreads();

    // ---- preload all B fragments into registers (loop-invariant) ----
    // B-fragment layout (16x16x32 bf16 WMMA): lane column n = ln16,
// elems j=0..7 hold K = kt*32 + half*8 + j, j=8..15 hold K = kt*32 + 16 + half*8 + (j-8).
    v16bf Bh[2][4][4];   // [slice][gate][kt]
    v16bf Bx[2][4];      // [slice][gate] : rows [W_ih^T | bias@K16 | 0]
    #pragma unroll
    for (int s = 0; s < 2; ++s) {
        const int hs = 2 * wq + s;
        #pragma unroll
        for (int gi = 0; gi < 4; ++gi) {
            const int gcol = gi * H_ + hs * 16 + ln16;
            #pragma unroll
            for (int kt = 0; kt < 4; ++kt) {
                const float* wp = W_hh + (size_t)gcol * H_ + kt * 32 + half * 8;
                v4f w0 = *(const v4f*)(wp + 0);
                v4f w1 = *(const v4f*)(wp + 4);
                v4f w2 = *(const v4f*)(wp + 16);
                v4f w3 = *(const v4f*)(wp + 20);
                v16bf f;
                #pragma unroll
                for (int j = 0; j < 4; ++j) {
                    f[j]      = (__bf16)w0[j];
                    f[4 + j]  = (__bf16)w1[j];
                    f[8 + j]  = (__bf16)w2[j];
                    f[12 + j] = (__bf16)w3[j];
                }
                Bh[s][gi][kt] = f;
            }
            v16bf fx;
            {
                const float* wp = W_ih + (size_t)gcol * IN_ + half * 8;
                v4f w0 = *(const v4f*)(wp + 0);
                v4f w1 = *(const v4f*)(wp + 4);
                #pragma unroll
                for (int j = 0; j < 4; ++j) {
                    fx[j]      = (__bf16)w0[j];
                    fx[4 + j]  = (__bf16)w1[j];
                }
                #pragma unroll
                for (int j = 8; j < 16; ++j) fx[j] = (__bf16)0.0f;
                if (half == 0) fx[8] = (__bf16)(b_ih[gcol] + b_hh[gcol]);
            }
            Bx[s][gi] = fx;
        }
    }

    // cell state for my two hidden slices (C-fragment layout:
    // elem r -> batch row M = r + 8*half, hidden col = hs*16 + ln16)
    v8f c[2];
    #pragma unroll
    for (int s = 0; s < 2; ++s)
        #pragma unroll
        for (int r = 0; r < 8; ++r) c[s][r] = 0.0f;

    const int brow = blockIdx.x * 16 + ln16;
    const float* xp = x + (size_t)brow * T_ * IN_ + half * 8;

    #pragma unroll 1
    for (int t = 0; t < T_; ++t) {
        const __bf16* hin  = &Hst[t & 1][0][0];
        __bf16*       hout = &Hst[(t + 1) & 1][0][0];

        // ---- A fragments for h_t (16x32 bf16 each) ----
        v16bf ah[4];
        #pragma unroll
        for (int kt = 0; kt < 4; ++kt) {
            const v4u* p0 = (const v4u*)&hin[ln16 * H_ + kt * 32 + half * 8];
            const v4u* p1 = (const v4u*)&hin[ln16 * H_ + kt * 32 + 16 + half * 8];
            ((v4u*)&ah[kt])[0] = *p0;   // elems 0..7  : K = kt*32 + (half?8:0)
            ((v4u*)&ah[kt])[1] = *p1;   // elems 8..15 : K = kt*32 + 16 + (half?8:0)
        }

        // ---- A fragment for [x_t | 1 | 0] (bias folded in at K==16) ----
        v4f x0 = *(const v4f*)(xp + 0);
        v4f x1 = *(const v4f*)(xp + 4);
        xp += IN_;
        __builtin_prefetch(xp + 8 * IN_, 0, 1);
        v16bf ax;
        #pragma unroll
        for (int i = 0; i < 4; ++i) { ax[i] = (__bf16)x0[i]; ax[4 + i] = (__bf16)x1[i]; }
        #pragma unroll
        for (int i = 8; i < 16; ++i) ax[i] = (__bf16)0.0f;
        ax[8] = (__bf16)(half ? 0.0f : 1.0f);

        // ---- gates for my two hidden slices (all operands in registers) ----
        #pragma unroll
        for (int s = 0; s < 2; ++s) {
            const int hs = 2 * wq + s;
            v8f acc[4];
            #pragma unroll
            for (int gi = 0; gi < 4; ++gi) {          // 0:i 1:f 2:g 3:o
                v8f a = {0.f, 0.f, 0.f, 0.f, 0.f, 0.f, 0.f, 0.f};
                #pragma unroll
                for (int kt = 0; kt < 4; ++kt)
                    a = __builtin_amdgcn_wmma_f32_16x16x32_bf16(
                            false, ah[kt], false, Bh[s][gi][kt], (short)0, a, false, false);
                a = __builtin_amdgcn_wmma_f32_16x16x32_bf16(
                        false, ax, false, Bx[s][gi], (short)0, a, false, false);
                acc[gi] = a;
            }
            // ---- elementwise LSTM update, write h_{t+1} slice ----
            #pragma unroll
            for (int r = 0; r < 8; ++r) {
                float ig = fast_sigm(acc[0][r]);
                float fg = fast_sigm(acc[1][r]);
                float gg = fast_tanh(acc[2][r]);
                float og = fast_sigm(acc[3][r]);
                float cv = fmaf(fg, c[s][r], ig * gg);
                c[s][r] = cv;
                float hv = og * fast_tanh(cv);
                const int m = r + half * 8;
                hout[m * H_ + hs * 16 + ln16] = (__bf16)hv;
            }
        }
        __syncthreads();   // h_{t+1} staging complete before next step reads it
    }

    // ---- final fc: out[b] = h_T . fc_w + fc_b (h_T is in buffer 0) ----
    if (tid < 16) {
        const __bf16* hr = &Hst[T_ & 1][tid][0];
        float s = 0.0f;
        #pragma unroll 4
        for (int k = 0; k < H_; ++k) s += (float)hr[k] * fc_w[k];
        out[blockIdx.x * 16 + tid] = s + fc_b[0];
    }
}

extern "C" void kernel_launch(void* const* d_in, const int* in_sizes, int n_in,
                              void* d_out, int out_size, void* d_ws, size_t ws_size,
                              hipStream_t stream) {
    const float* x    = (const float*)d_in[0];
    const float* W_ih = (const float*)d_in[1];
    const float* W_hh = (const float*)d_in[2];
    const float* b_ih = (const float*)d_in[3];
    const float* b_hh = (const float*)d_in[4];
    const float* fc_w = (const float*)d_in[5];
    const float* fc_b = (const float*)d_in[6];
    float* out = (float*)d_out;

    slstm_wmma_kernel<<<dim3(64), dim3(128), 0, stream>>>(
        x, W_ih, W_hh, b_ih, b_hh, fc_w, fc_b, out);
}
